// BiLSTM_CRF_32452772888794
// MI455X (gfx1250) — matrix operand order repaired and checked by reading, not compiled, not run on hardware
//
#include <hip/hip_runtime.h>
#include <hip/hip_bf16.h>

// ---------------- problem constants (from reference) ----------------
#define TT   4096      // sequence length
#define EE   300       // embedding dim
#define EP   320       // E padded to multiple of 32 (WMMA bf16 K-step)
#define HD   128       // per-direction hidden
#define G4   512       // 4*HD gate width
#define HH   256       // concat hidden
#define KK   11        // number of tags
#define START_TAG 9
#define STOP_TAG  10

typedef __bf16 bf16;
typedef __attribute__((ext_vector_type(16))) bf16  v16bf;
typedef __attribute__((ext_vector_type(8)))  bf16  v8bf;
typedef __attribute__((ext_vector_type(8)))  float v8f;

// float -> bf16 (round-to-nearest-even), via bit punning
__device__ __forceinline__ bf16 f2bf(float f) {
    union { float f; unsigned u; } in; in.f = f;
    unsigned u = in.u + 0x7fffu + ((in.u >> 16) & 1u);
    union { unsigned short s; bf16 b; } out; out.s = (unsigned short)(u >> 16);
    return out.b;
}
__device__ __forceinline__ float sigf(float x) { return 1.f / (1.f + expf(-x)); }

// ---------------- kernel 1: embedding gather -> bf16 (padded) ----------------
__global__ void k_embed(const int* __restrict__ inputs,
                        const float* __restrict__ word_embed,
                        bf16* __restrict__ emb) {
    int t = blockIdx.x;           // 0..T-1
    int c = threadIdx.x;          // 0..EP-1
    float v = 0.f;
    if (c < EE) v = word_embed[(size_t)inputs[t] * EE + c];
    emb[(size_t)t * EP + c] = f2bf(v);
}

// ---------------- kernel 2: Wih (both dirs) -> bf16 (padded) ----------------
__global__ void k_cvt_wih(const float* __restrict__ wih_f,
                          const float* __restrict__ wih_b,
                          bf16* __restrict__ dst) {
    int row = blockIdx.x;         // 0..511
    int dir = blockIdx.y;         // 0/1
    int c   = threadIdx.x;        // 0..EP-1
    const float* src = dir ? wih_b : wih_f;
    float v = (c < EE) ? src[(size_t)row * EE + c] : 0.f;
    dst[((size_t)dir * G4 + row) * EP + c] = f2bf(v);
}

// ---------------- kernel 3: Whh (both dirs) -> bf16 ----------------
__global__ void k_cvt_whh(const float* __restrict__ whh_f,
                          const float* __restrict__ whh_b,
                          bf16* __restrict__ dst) {
    int row = blockIdx.x;         // 0..511
    int dir = blockIdx.y;
    int c   = threadIdx.x;        // 0..127
    const float* src = dir ? whh_b : whh_f;
    dst[((size_t)dir * G4 + row) * HD + c] = f2bf(src[(size_t)row * HD + c]);
}

// ---------------- kernel 4: xW = emb @ Wih^T + (bih+bhh), WMMA bf16 ----------
// grid (T/16, G4/16, 2), block 32 (one wave per 16x16 tile)
__global__ void k_gemm_xw(const bf16* __restrict__ emb,
                          const bf16* __restrict__ wih,
                          const float* __restrict__ bih_f,
                          const float* __restrict__ bhh_f,
                          const float* __restrict__ bih_b,
                          const float* __restrict__ bhh_b,
                          float* __restrict__ xw) {
    const int mt   = blockIdx.x * 16;
    const int nt   = blockIdx.y * 16;
    const int dir  = blockIdx.z;
    const int lane = threadIdx.x;          // 0..31, wave32
    const int m    = mt + (lane & 15);
    const int half = lane >> 4;            // 0: K 0-15; 1: K 16-31
    const int n    = nt + (lane & 15);

    const bf16* B = wih + (size_t)dir * G4 * EP;

    v8f acc;
#pragma unroll
    for (int r = 0; r < 8; ++r) acc[r] = 0.f;

#pragma unroll
    for (int kb = 0; kb < EP / 32; ++kb) {
        // A fragment: row m, 16 contiguous K values per lane half (same order as B)
        const int ka = kb * 32 + half * 16;
        v8bf alo = *(const v8bf*)(emb + (size_t)m * EP + ka);
        v8bf ahi = *(const v8bf*)(emb + (size_t)m * EP + ka + 8);
        v16bf a = __builtin_shufflevector(alo, ahi,
                   0,1,2,3,4,5,6,7,8,9,10,11,12,13,14,15);
        // B fragment: 32x16, lane = column N, 16 contiguous K values
        const int kbb = kb * 32 + half * 16;
        v16bf b = *(const v16bf*)(B + (size_t)n * EP + kbb);

        acc = __builtin_amdgcn_wmma_f32_16x16x32_bf16(
                  false, a, false, b, (short)0, acc, false, false);
    }

    const float bias = dir ? (bih_b[n] + bhh_b[n]) : (bih_f[n] + bhh_f[n]);
    float* out = xw + (size_t)dir * TT * G4;
#pragma unroll
    for (int r = 0; r < 8; ++r) {
        int row = mt + r + half * 8;       // C/D layout (ISA 7.12.2)
        out[(size_t)row * G4 + n] = acc[r] + bias;
    }
}

// ---------------- kernel 5: sequential LSTM recurrence (per direction) ------
// grid 2 (dir), block 512 = 16 waves.
// Per step: z = Whh x h done with WMMA: A (Whh tiles) preloaded in registers,
// B = h (bf16, in LDS) replicated across all 16 columns; only column 0
// (lanes 0/16) of each D tile is kept. Gate math on threads 0..127.
__global__ void k_recur(const float* __restrict__ xw,
                        const bf16* __restrict__ whh,
                        float* __restrict__ hs) {
    const int dir  = blockIdx.x;
    const int t    = threadIdx.x;          // 0..511
    const int lane = t & 31;
    const int wv   = t >> 5;               // wave id, 0..15
    const int half = lane >> 4;

    __shared__ __align__(32) bf16 hbf[HD]; // h(t-1), bf16, B-operand source
    __shared__ float z[G4];                // Whh x h results

    // Preload this wave's 2 M-tiles x 4 K-blocks of Whh into registers
    // (step-invariant A fragments: 64 VGPRs).
    const bf16* W = whh + (size_t)dir * G4 * HD;
    v16bf afrag[2][4];
#pragma unroll
    for (int i = 0; i < 2; ++i) {
        const int m = wv * 32 + i * 16 + (lane & 15);
#pragma unroll
        for (int kb = 0; kb < 4; ++kb) {
            const int ka = kb * 32 + half * 16;
            v8bf lo = *(const v8bf*)(W + (size_t)m * HD + ka);
            v8bf hi = *(const v8bf*)(W + (size_t)m * HD + ka + 8);
            afrag[i][kb] = __builtin_shufflevector(lo, hi,
                0,1,2,3,4,5,6,7,8,9,10,11,12,13,14,15);
        }
    }

    const float* xwd = xw + (size_t)dir * TT * G4;
    float*       hsd = hs + (size_t)dir * TT * HD;

    if (t < HD) hbf[t] = f2bf(0.f);
    float c = 0.f;
    __syncthreads();

    for (int s = 0; s < TT; ++s) {
        const int tt = dir ? (TT - 1 - s) : s;
        if (s + 1 < TT) {
            const int tn = dir ? (TT - 2 - s) : (s + 1);
            __builtin_prefetch(&xwd[(size_t)tn * G4 + t]);  // global_prefetch_b8
        }

        // B fragments: the lanes of each half read the same 16 bf16 of h (LDS broadcast);
        // all 16 columns of B equal h, so every D column holds Whh x h.
        v16bf bfrag[4];
#pragma unroll
        for (int kb = 0; kb < 4; ++kb)
            bfrag[kb] = *(const v16bf*)(&hbf[kb * 32 + half * 16]);

#pragma unroll
        for (int i = 0; i < 2; ++i) {
            v8f acc;
#pragma unroll
            for (int r = 0; r < 8; ++r) acc[r] = 0.f;
#pragma unroll
            for (int kb = 0; kb < 4; ++kb)
                acc = __builtin_amdgcn_wmma_f32_16x16x32_bf16(
                          false, afrag[i][kb], false, bfrag[kb],
                          (short)0, acc, false, false);
            // keep column N=0: lanes 0 (rows r) and 16 (rows 8+r)
            if ((lane & 15) == 0) {
                const int rowb = wv * 32 + i * 16 + half * 8;
#pragma unroll
                for (int r = 0; r < 8; ++r) z[rowb + r] = acc[r];
            }
        }
        __syncthreads();

        if (t < HD) {
            const float* xrow = &xwd[(size_t)tt * G4];
            const float ig = sigf (z[t]          + xrow[t]);
            const float fg = sigf (z[HD + t]     + xrow[HD + t]);
            const float gg = tanhf(z[2 * HD + t] + xrow[2 * HD + t]);
            const float og = sigf (z[3 * HD + t] + xrow[3 * HD + t]);
            c = fg * c + ig * gg;
            const float hv = og * tanhf(c);
            hbf[t] = f2bf(hv);
            hsd[(size_t)tt * HD + t] = hv;
        }
        __syncthreads();
    }
}

// ---------------- kernel 6: feats = [h_f, h_b] @ fc_W^T + fc_b --------------
__global__ void k_feats(const float* __restrict__ hs,
                        const float* __restrict__ fc_w,
                        const float* __restrict__ fc_b,
                        float* __restrict__ feats) {
    const int idx = blockIdx.x * blockDim.x + threadIdx.x;
    if (idx >= TT * KK) return;
    const int t = idx / KK, k = idx % KK;
    const float* wf = fc_w + (size_t)k * HH;
    const float* hf = hs + (size_t)t * HD;
    const float* hb = hs + (size_t)TT * HD + (size_t)t * HD;
    float s = fc_b[k];
#pragma unroll 4
    for (int j = 0; j < HD; ++j) s = fmaf(wf[j], hf[j], s);
#pragma unroll 4
    for (int j = 0; j < HD; ++j) s = fmaf(wf[HD + j], hb[j], s);
    feats[idx] = s;
}

// ---------------- kernel 7: CRF forward scan + gold score (1 wave) ----------
__global__ void k_crf(const float* __restrict__ feats,
                      const int* __restrict__ tags,
                      const float* __restrict__ trans,
                      float* __restrict__ out) {
    const int lane = threadIdx.x;                 // 0..31 (wave32)
    const int j = (lane < KK) ? lane : (KK - 1);  // clamp extra lanes

    float tcol[KK];                               // trans[i][j]
#pragma unroll
    for (int i = 0; i < KK; ++i) tcol[i] = trans[i * KK + j];

    float alpha = (j == START_TAG) ? 0.f : -1000.f;

    for (int t = 0; t < TT; ++t) {
        float v[KK];
        float mx = -3.4e38f;
#pragma unroll
        for (int i = 0; i < KK; ++i) {
            const float ai = __shfl(alpha, i, 32);
            const float s = ai + tcol[i];
            v[i] = s;
            mx = fmaxf(mx, s);
        }
        float ss = 0.f;
#pragma unroll
        for (int i = 0; i < KK; ++i) ss += expf(v[i] - mx);
        alpha = mx + logf(ss) + feats[(size_t)t * KK + j];
    }

    // forward_score = logsumexp(alpha + trans[:, STOP])
    const float fin = alpha + trans[j * KK + STOP_TAG];
    float mx = -3.4e38f;
#pragma unroll
    for (int i = 0; i < KK; ++i) mx = fmaxf(mx, __shfl(fin, i, 32));
    float ss = 0.f;
#pragma unroll
    for (int i = 0; i < KK; ++i) ss += expf(__shfl(fin, i, 32) - mx);
    const float forward_score = mx + logf(ss);

    // gold path score: lane-strided partial sums
    float part = 0.f;
    for (int t = lane; t < TT; t += 32)
        part += feats[(size_t)t * KK + tags[t]];
    for (int t = lane; t <= TT; t += 32) {
        const int a = (t == 0) ? START_TAG : tags[t - 1];
        const int b = (t < TT) ? tags[t] : START_TAG;
        part += trans[a * KK + b];
    }
#pragma unroll
    for (int off = 16; off > 0; off >>= 1)
        part += __shfl_down(part, off, 32);
    if (lane == 0) out[0] = forward_score - part;
}

// ---------------- host launcher ----------------
extern "C" void kernel_launch(void* const* d_in, const int* in_sizes, int n_in,
                              void* d_out, int out_size, void* d_ws, size_t ws_size,
                              hipStream_t stream) {
    (void)in_sizes; (void)n_in; (void)out_size; (void)ws_size;

    const int*   inputs     = (const int*)  d_in[0];
    const int*   tags       = (const int*)  d_in[1];
    const float* word_embed = (const float*)d_in[2];
    const float* wih_f      = (const float*)d_in[3];
    const float* whh_f      = (const float*)d_in[4];
    const float* bih_f      = (const float*)d_in[5];
    const float* bhh_f      = (const float*)d_in[6];
    const float* wih_b      = (const float*)d_in[7];
    const float* whh_b      = (const float*)d_in[8];
    const float* bih_b      = (const float*)d_in[9];
    const float* bhh_b      = (const float*)d_in[10];
    const float* fc_w       = (const float*)d_in[11];
    const float* fc_b       = (const float*)d_in[12];
    const float* trans      = (const float*)d_in[13];

    // workspace layout (all sizes are multiples of 256 bytes)
    char* w = (char*)d_ws;
    bf16*  emb   = (bf16*)w;          w += (size_t)TT * EP * sizeof(bf16);      // 2.5 MB
    bf16*  wihb  = (bf16*)w;          w += (size_t)2 * G4 * EP * sizeof(bf16);  // 640 KB
    bf16*  whhb  = (bf16*)w;          w += (size_t)2 * G4 * HD * sizeof(bf16);  // 256 KB
    float* xw    = (float*)w;         w += (size_t)2 * TT * G4 * sizeof(float); // 16 MB
    float* hs    = (float*)w;         w += (size_t)2 * TT * HD * sizeof(float); // 4 MB
    float* feats = (float*)w;         w += (size_t)TT * KK * sizeof(float);     // 176 KB

    k_embed  <<<TT, EP, 0, stream>>>(inputs, word_embed, emb);
    k_cvt_wih<<<dim3(G4, 2), EP, 0, stream>>>(wih_f, wih_b, wihb);
    k_cvt_whh<<<dim3(G4, 2), HD, 0, stream>>>(whh_f, whh_b, whhb);

    k_gemm_xw<<<dim3(TT / 16, G4 / 16, 2), 32, 0, stream>>>(
        emb, wihb, bih_f, bhh_f, bih_b, bhh_b, xw);

    k_recur<<<2, G4, 0, stream>>>(xw, whhb, hs);

    k_feats<<<(TT * KK + 255) / 256, 256, 0, stream>>>(hs, fc_w, fc_b, feats);

    k_crf<<<1, 32, 0, stream>>>(feats, tags, trans, (float*)d_out);
}
